// MultiHeadAttention_48284022342253
// MI455X (gfx1250) — matrix operand order repaired and checked by reading, not compiled, not run on hardware
//
#include <hip/hip_runtime.h>
#include <hip/hip_bf16.h>
#include <math.h>

// ---------------------------------------------------------------------------
// MI455X (gfx1250) multi-head attention, bf16 WMMA path, v2.
// Changes vs v1: 32-row M-tiles everywhere so every B-fragment load feeds two
// v_wmma_f32_16x16x32_bf16 ops (2x operand reuse, halves L2 traffic per FLOP).
//   x:[4,2048,768] f32; Wq/Wk/Wv/Wo:[768,768] f32 (out = in @ W.T + b)
//   scores = QK^T / 64, softmax, @V, merge heads, @Wo.T + bo  -> f32 out
// ---------------------------------------------------------------------------

typedef __attribute__((ext_vector_type(16))) __bf16        v16bf;
typedef __attribute__((ext_vector_type(8)))  float         v8f;
typedef __attribute__((ext_vector_type(8)))  unsigned int  v8u;

#define D_MODEL 768
#define N_HEADS 12
#define DK      64
#define SEQ     2048
#define BATCH   4
#define MROWS   (BATCH * SEQ) /* 8192 */

__device__ __forceinline__ unsigned short f32_to_bf16(float f) {
  unsigned int u = __float_as_uint(f);
  u += 0x7FFFu + ((u >> 16) & 1u);   // round-to-nearest-even
  return (unsigned short)(u >> 16);
}

__device__ __forceinline__ v8f wmma_bf16(v16bf a, v16bf b, v8f c) {
  // v_wmma_f32_16x16x32_bf16 : D = A(16x32) * B(32x16) + C(16x16 f32)
  return __builtin_amdgcn_wmma_f32_16x16x32_bf16(false, a, false, b,
                                                 (short)0, c, false, false);
}

// A-fragment: 16(M)x32(K) bf16 from row-major src[row][k], tile at (row0,k0).
// ISA layout: lane m (0-15) holds K = {0..7, 16..23}; lane m+16 holds {8..15, 24..31}.
__device__ __forceinline__ v16bf load_A(const unsigned short* __restrict__ src,
                                        int ld, int row0, int k0, int lane) {
  const int row = row0 + (lane & 15);
  const int hk  = (lane >> 4) << 3;                 // 0 or 8
  const unsigned short* p = src + (size_t)row * ld + k0 + hk;
  const uint4 lo = *(const uint4*)(p);              // K = k0+hk    .. +7
  const uint4 hi = *(const uint4*)(p + 16);         // K = k0+16+hk .. +7
  v8u u; u[0]=lo.x; u[1]=lo.y; u[2]=lo.z; u[3]=lo.w;
         u[4]=hi.x; u[5]=hi.y; u[6]=hi.z; u[7]=hi.w;
  return __builtin_bit_cast(v16bf, u);
}

// B-fragment: 32(K)x16(N) bf16, column-major B == row-major w[n][k].
// ISA layout: lane n (0-15) holds K = {0..7, 16..23}; lane n+16 holds {8..15, 24..31}.
__device__ __forceinline__ v16bf load_B(const unsigned short* __restrict__ w,
                                        int ld, int n0, int k0, int lane) {
  const int col = n0 + (lane & 15);
  const int kh  = (lane >> 4) << 3;                 // 0 or 8
  const unsigned short* p = w + (size_t)col * ld + k0 + kh;
  const uint4 lo = *(const uint4*)(p);
  const uint4 hi = *(const uint4*)(p + 16);
  v8u u; u[0]=lo.x; u[1]=lo.y; u[2]=lo.z; u[3]=lo.w;
         u[4]=hi.x; u[5]=hi.y; u[6]=hi.z; u[7]=hi.w;
  return __builtin_bit_cast(v16bf, u);
}

// ---------------------------------------------------------------------------
__global__ void cvt_f32_bf16_kernel(const float* __restrict__ src,
                                    unsigned short* __restrict__ dst, int n) {
  int i = blockIdx.x * blockDim.x + threadIdx.x;
  if (i < n) dst[i] = f32_to_bf16(src[i]);
}

// ---------------------------------------------------------------------------
// Q/K/V projection: out = x @ W.T + b (bf16 in, f32 accum, bf16 out).
// One wave per 32(M) x 64(N) strip; every B-frag feeds two WMMAs.
// v_transposed=1 stores [B,H,64,S] (for V).
__global__ void proj_qkv_kernel(const unsigned short* __restrict__ xbf,
                                const unsigned short* __restrict__ wbf,
                                const float* __restrict__ bias,
                                unsigned short* __restrict__ dst,
                                int v_transposed) {
  const int lane = threadIdx.x & 31;
  const int m0 = blockIdx.x * 32;
  const int n0 = blockIdx.y * 64;
  v8f acc[2][4] = {};
  for (int k0 = 0; k0 < D_MODEL; k0 += 32) {
    const v16bf a0 = load_A(xbf, D_MODEL, m0,      k0, lane);
    const v16bf a1 = load_A(xbf, D_MODEL, m0 + 16, k0, lane);
#pragma unroll
    for (int t = 0; t < 4; ++t) {
      const v16bf bf = load_B(wbf, D_MODEL, n0 + t * 16, k0, lane);
      acc[0][t] = wmma_bf16(a0, bf, acc[0][t]);
      acc[1][t] = wmma_bf16(a1, bf, acc[1][t]);
    }
  }
  // C layout: lane 0-15 -> M=r, lane 16-31 -> M=8+r; N = lane&15
  const int ncol = lane & 15;
  const int mh   = (lane >> 4) << 3;
#pragma unroll
  for (int g = 0; g < 2; ++g) {
#pragma unroll
    for (int t = 0; t < 4; ++t) {
      const int n = n0 + t * 16 + ncol;
      const float bv = bias[n];
      const int h = n >> 6, d = n & 63;
#pragma unroll
      for (int r = 0; r < 8; ++r) {
        const int sg = m0 + g * 16 + mh + r;   // global row in [0,8192)
        const int bb = sg >> 11, sl = sg & 2047;
        const unsigned short val = f32_to_bf16(acc[g][t][r] + bv);
        if (v_transposed)
          dst[((size_t)(bb * N_HEADS + h) * DK + d) * SEQ + sl] = val;
        else
          dst[((size_t)(bb * N_HEADS + h) * SEQ + sl) * DK + d] = val;
      }
    }
  }
}

// ---------------------------------------------------------------------------
// Flash attention: one wave handles 32 query rows of one (b,h).
// K/V fragments are loaded once per 32-key step and reused for both q-groups.
__global__ void flash_attn_kernel(const unsigned short* __restrict__ Q,
                                  const unsigned short* __restrict__ K,
                                  const unsigned short* __restrict__ Vt,
                                  unsigned short* __restrict__ att) {
  __shared__ unsigned short pbuf[32 * 36];          // P transpose scratch (+pad)
  const int lane = threadIdx.x & 31;
  const int q0 = blockIdx.x * 32;
  const int h  = blockIdx.y;
  const int b  = blockIdx.z;
  const unsigned short* Qh = Q  + (size_t)(b * N_HEADS + h) * SEQ * DK;
  const unsigned short* Kh = K  + (size_t)(b * N_HEADS + h) * SEQ * DK;
  const unsigned short* Vh = Vt + (size_t)(b * N_HEADS + h) * DK * SEQ;

  // Q fragments are loop-invariant: [group g][d-step] (d_k = 64 -> 2 K-steps)
  v16bf qa[2][2];
#pragma unroll
  for (int g = 0; g < 2; ++g) {
    qa[g][0] = load_A(Qh, DK, q0 + g * 16, 0,  lane);
    qa[g][1] = load_A(Qh, DK, q0 + g * 16, 32, lane);
  }

  v8f o[2][4] = {};                                 // O accum: 2 x 16(M) x 64(d)
  float mrow[2][8], lrow[2][8];
#pragma unroll
  for (int g = 0; g < 2; ++g)
#pragma unroll
    for (int r = 0; r < 8; ++r) { mrow[g][r] = -1e30f; lrow[g][r] = 0.0f; }
  const float scale = 1.0f / 64.0f;                 // reference divides by d_k

  for (int kb = 0; kb < SEQ; kb += 32) {
    __builtin_prefetch(Kh + (size_t)(kb + 32) * DK, 0, 0);
    __builtin_prefetch(Vh + (size_t)kb + 32, 0, 0);

    // scores tile S = Q * K^T : 32(q) x 32(k) = 4 C tiles; K frags shared by groups
    v8f s[2][2] = {};
    {
      const v16bf k00 = load_B(Kh, DK, kb,      0,  lane);
      const v16bf k01 = load_B(Kh, DK, kb,      32, lane);
      const v16bf k10 = load_B(Kh, DK, kb + 16, 0,  lane);
      const v16bf k11 = load_B(Kh, DK, kb + 16, 32, lane);
#pragma unroll
      for (int g = 0; g < 2; ++g) {
        s[g][0] = wmma_bf16(qa[g][0], k00, s[g][0]);
        s[g][0] = wmma_bf16(qa[g][1], k01, s[g][0]);
        s[g][1] = wmma_bf16(qa[g][0], k10, s[g][1]);
        s[g][1] = wmma_bf16(qa[g][1], k11, s[g][1]);
      }
    }

    // online softmax per q-group, in C layout: per-row stats via 16-lane
    // butterflies (xor masks 1,2,4,8 never cross the lane-half boundary)
#pragma unroll
    for (int g = 0; g < 2; ++g) {
      float alpha[8];
#pragma unroll
      for (int r = 0; r < 8; ++r) {
        float a0 = s[g][0][r] * scale, a1 = s[g][1][r] * scale;
        float pm = fmaxf(a0, a1);
#pragma unroll
        for (int msk = 1; msk <= 8; msk <<= 1)
          pm = fmaxf(pm, __shfl_xor(pm, msk, 32));
        const float mn = fmaxf(mrow[g][r], pm);
        alpha[r] = __expf(mrow[g][r] - mn);
        a0 = __expf(a0 - mn);
        a1 = __expf(a1 - mn);
        float ps = a0 + a1;
#pragma unroll
        for (int msk = 1; msk <= 8; msk <<= 1)
          ps += __shfl_xor(ps, msk, 32);
        lrow[g][r] = lrow[g][r] * alpha[r] + ps;
        mrow[g][r] = mn;
        s[g][0][r] = a0; s[g][1][r] = a1;
      }
#pragma unroll
      for (int t = 0; t < 4; ++t)
#pragma unroll
        for (int r = 0; r < 8; ++r) o[g][t][r] *= alpha[r];
    }

    // transpose P: C layout (lane=k-col) -> A layout (lane=q-row) via LDS
    {
      const int n = lane & 15, hm = (lane >> 4) << 3;
#pragma unroll
      for (int g = 0; g < 2; ++g)
#pragma unroll
        for (int r = 0; r < 8; ++r) {
          pbuf[(g * 16 + hm + r) * 36 + n]      = f32_to_bf16(s[g][0][r]);
          pbuf[(g * 16 + hm + r) * 36 + 16 + n] = f32_to_bf16(s[g][1][r]);
        }
    }
    v16bf pa[2];
    {
      const int row = lane & 15;
      const int hk  = (lane >> 4) << 3;
      const unsigned int* pb32 = (const unsigned int*)pbuf;
#pragma unroll
      for (int g = 0; g < 2; ++g) {
        const int b0 = ((g * 16 + row) * 36 + hk) >> 1;       // even u16 offsets
        const int b1 = ((g * 16 + row) * 36 + 16 + hk) >> 1;
        v8u u;
#pragma unroll
        for (int j = 0; j < 4; ++j) { u[j] = pb32[b0 + j]; u[4 + j] = pb32[b1 + j]; }
        pa[g] = __builtin_bit_cast(v16bf, u);
      }
    }

    // O += P(32x32) @ V(32x64); V frags loaded once, used by both q-groups
#pragma unroll
    for (int t = 0; t < 4; ++t) {
      const v16bf vb = load_B(Vh, SEQ, t * 16, kb, lane);
      o[0][t] = wmma_bf16(pa[0], vb, o[0][t]);
      o[1][t] = wmma_bf16(pa[1], vb, o[1][t]);
    }
  }

  // epilogue: normalize and store merged-head bf16 [B,S,768]
  const int ncol = lane & 15, mh = (lane >> 4) << 3;
#pragma unroll
  for (int g = 0; g < 2; ++g)
#pragma unroll
    for (int t = 0; t < 4; ++t) {
      const int d = t * 16 + ncol;
#pragma unroll
      for (int r = 0; r < 8; ++r) {
        const int s = q0 + g * 16 + mh + r;
        att[((size_t)(b * SEQ + s)) * D_MODEL + h * DK + d] =
            f32_to_bf16(o[g][t][r] / lrow[g][r]);
      }
    }
}

// ---------------------------------------------------------------------------
// Output projection: out = att @ Wo.T + bo, fp32 result. 32(M) x 64(N) per wave.
__global__ void proj_out_kernel(const unsigned short* __restrict__ attbf,
                                const unsigned short* __restrict__ wbf,
                                const float* __restrict__ bias,
                                float* __restrict__ out) {
  const int lane = threadIdx.x & 31;
  const int m0 = blockIdx.x * 32;
  const int n0 = blockIdx.y * 64;
  v8f acc[2][4] = {};
  for (int k0 = 0; k0 < D_MODEL; k0 += 32) {
    const v16bf a0 = load_A(attbf, D_MODEL, m0,      k0, lane);
    const v16bf a1 = load_A(attbf, D_MODEL, m0 + 16, k0, lane);
#pragma unroll
    for (int t = 0; t < 4; ++t) {
      const v16bf bf = load_B(wbf, D_MODEL, n0 + t * 16, k0, lane);
      acc[0][t] = wmma_bf16(a0, bf, acc[0][t]);
      acc[1][t] = wmma_bf16(a1, bf, acc[1][t]);
    }
  }
  const int ncol = lane & 15;
  const int mh   = (lane >> 4) << 3;
#pragma unroll
  for (int g = 0; g < 2; ++g)
#pragma unroll
    for (int t = 0; t < 4; ++t) {
      const int n = n0 + t * 16 + ncol;
      const float bv = bias[n];
#pragma unroll
      for (int r = 0; r < 8; ++r) {
        const int sg = m0 + g * 16 + mh + r;
        out[(size_t)sg * D_MODEL + n] = acc[g][t][r] + bv;
      }
    }
}

// ---------------------------------------------------------------------------
extern "C" void kernel_launch(void* const* d_in, const int* in_sizes, int n_in,
                              void* d_out, int out_size, void* d_ws, size_t ws_size,
                              hipStream_t stream) {
  const float* x  = (const float*)d_in[0];
  const float* Wq = (const float*)d_in[1];
  const float* bq = (const float*)d_in[2];
  const float* Wk = (const float*)d_in[3];
  const float* bk = (const float*)d_in[4];
  const float* Wv = (const float*)d_in[5];
  const float* bv = (const float*)d_in[6];
  const float* Wo = (const float*)d_in[7];
  const float* bo = (const float*)d_in[8];
  float* out = (float*)d_out;

  char* ws = (char*)d_ws;
  size_t off = 0;
  const size_t NX   = (size_t)MROWS * D_MODEL;        // 6,291,456
  const size_t NW   = (size_t)D_MODEL * D_MODEL;      //   589,824
  const size_t NQKV = (size_t)BATCH * N_HEADS * SEQ * DK;

  unsigned short* xbf  = (unsigned short*)(ws + off); off += NX  * 2;
  unsigned short* wqbf = (unsigned short*)(ws + off); off += NW  * 2;
  unsigned short* wkbf = (unsigned short*)(ws + off); off += NW  * 2;
  unsigned short* wvbf = (unsigned short*)(ws + off); off += NW  * 2;
  unsigned short* wobf = (unsigned short*)(ws + off); off += NW  * 2;
  unsigned short* Qbf  = (unsigned short*)(ws + off); off += NQKV * 2;
  unsigned short* Kbf  = (unsigned short*)(ws + off); off += NQKV * 2;
  unsigned short* Vtbf = (unsigned short*)(ws + off); off += NQKV * 2;
  unsigned short* attb = (unsigned short*)(ws + off); off += NX  * 2;

  // 1) fp32 -> bf16 conversions
  cvt_f32_bf16_kernel<<<dim3((unsigned)((NX + 255) / 256)), 256, 0, stream>>>(x, xbf, (int)NX);
  cvt_f32_bf16_kernel<<<dim3((unsigned)((NW + 255) / 256)), 256, 0, stream>>>(Wq, wqbf, (int)NW);
  cvt_f32_bf16_kernel<<<dim3((unsigned)((NW + 255) / 256)), 256, 0, stream>>>(Wk, wkbf, (int)NW);
  cvt_f32_bf16_kernel<<<dim3((unsigned)((NW + 255) / 256)), 256, 0, stream>>>(Wv, wvbf, (int)NW);
  cvt_f32_bf16_kernel<<<dim3((unsigned)((NW + 255) / 256)), 256, 0, stream>>>(Wo, wobf, (int)NW);

  // 2) Q/K/V projections (one wave per 32x64 output strip)
  dim3 pg(MROWS / 32, D_MODEL / 64);                  // 256 x 12
  proj_qkv_kernel<<<pg, 32, 0, stream>>>(xbf, wqbf, bq, Qbf, 0);
  proj_qkv_kernel<<<pg, 32, 0, stream>>>(xbf, wkbf, bk, Kbf, 0);
  proj_qkv_kernel<<<pg, 32, 0, stream>>>(xbf, wvbf, bv, Vtbf, 1);

  // 3) flash attention: one wave per 32 query rows per (b,h)
  dim3 fg(SEQ / 32, N_HEADS, BATCH);                  // 64 x 12 x 4
  flash_attn_kernel<<<fg, 32, 0, stream>>>(Qbf, Kbf, Vtbf, attb);

  // 4) output projection (fp32 out)
  proj_out_kernel<<<pg, 32, 0, stream>>>(attb, wobf, bo, out);
}
